// Global_Reactivity_Attention_10943576670418
// MI455X (gfx1250) — compile-verified, matrix operand order checked by reading, not run on hardware
//
#include <hip/hip_runtime.h>
#include <hip/hip_bf16.h>
#include <math.h>

typedef __attribute__((ext_vector_type(16))) _Float16 v16h;
typedef __attribute__((ext_vector_type(8)))  _Float16 v8h;
typedef __attribute__((ext_vector_type(8)))  float    v8f;
typedef int v4i __attribute__((vector_size(16)));   // matches builtin param type

#define DMODEL 1024
#define NHEAD  16
#define NLAYER 4
#define BATCH  2
#define SEQ    1024
#define DK     64
#define DFF    2048
#define MROWS  (BATCH * SEQ)   // 2048

// CDNA5 async global->LDS path (ASYNCcnt-tracked, bypasses VGPRs).
// Guarded so an older toolchain silently falls back to the sync path.
#if defined(__HIP_DEVICE_COMPILE__) && \
    __has_builtin(__builtin_amdgcn_global_load_async_to_lds_b128) && \
    __has_builtin(__builtin_amdgcn_s_wait_asynccnt)
#define ASYNC_LDS 1
#else
#define ASYNC_LDS 0
#endif

#define TO_GLB(p) ((__attribute__((address_space(1))) v4i*)(void*)(p))
#define TO_LDS(p) ((__attribute__((address_space(3))) v4i*)(void*)(p))

// ---------------------------------------------------------------------------
// helpers
// ---------------------------------------------------------------------------
__device__ __forceinline__ v8f wmma_f16(v16h a, v16h b, v8f c) {
  // D = A(16x32 f16) * B(32x16 f16) + C(16x16 f32)
  return __builtin_amdgcn_wmma_f32_16x16x32_f16(
      /*neg_a=*/false, a, /*neg_b=*/false, b,
      /*c_mod=*/(short)0, c, /*reuse_a=*/false, /*reuse_b=*/false);
}

__device__ __forceinline__ int frag_k(int v, int p, int half) {
  // CDNA5 16-bit A-matrix 16x32 layout: VGPR v, packed half p, lane-half
  return ((v < 4) ? 0 : 16) + 8 * half + 2 * (v & 3) + p;
}

__device__ __forceinline__ float gelu_f(float x) {
  float x3 = x * x * x;
  return 0.5f * x * (1.0f + tanhf(0.79788456080286535588f * (x + 0.044715f * x3)));
}

// ---------------------------------------------------------------------------
// f32 -> f16 bulk conversion (one 8-wide vector per thread)
// ---------------------------------------------------------------------------
__global__ __launch_bounds__(256)
void cvt_kernel(const float* __restrict__ in, _Float16* __restrict__ out) {
  size_t i = ((size_t)blockIdx.x * 256 + threadIdx.x) * 8;
  const float* p = in + i;
  v8h o;
#pragma unroll
  for (int j = 0; j < 8; ++j) o[j] = (_Float16)p[j];
  *(v8h*)(out + i) = o;
}

// ---------------------------------------------------------------------------
// f16 GEMM: C[M,N] = act(A[M,K] * W[K,N] + bias[N])
// block tile 128x64, k-step 32, 256 threads = 8 waves (4x2 wave grid),
// each wave owns a 32x32 register tile -> 4 WMMAs per k-step.
// Double-buffered LDS; async global->LDS copy when the toolchain has it.
// ---------------------------------------------------------------------------
template <int ACT, int OUT16>   // ACT: 0=none,1=gelu ; OUT16: store f16 or f32
__global__ __launch_bounds__(256)
void gemm16_kernel(const _Float16* __restrict__ A, const _Float16* __restrict__ W,
                   const float* __restrict__ bias, void* __restrict__ Cout,
                   int M, int N, int K) {
  __shared__ _Float16 As[2][128][32] __attribute__((aligned(16)));
  __shared__ _Float16 Bs[2][32][64]  __attribute__((aligned(16)));

  const int tid  = threadIdx.x;
  const int lane = tid & 31;
  const int wv   = tid >> 5;
  const int wm   = wv >> 1;       // 0..3 : 32-row slice
  const int wn   = wv & 1;        // 0..1 : 32-col slice
  const int half = lane >> 4;
  const int mr   = lane & 15;
  const int m0   = blockIdx.y * 128;
  const int n0   = blockIdx.x * 64;

  // cooperative-load coordinates (A: 4096 halfs = 2 x b128/thread, B: 2048 halfs)
  const int a0 = tid * 8,        ar0 = a0 >> 5, ac0 = a0 & 31;
  const int a1 = tid * 8 + 2048, ar1 = a1 >> 5, ac1 = a1 & 31;
  const int b0 = tid * 8,        br0 = b0 >> 6, bc0 = b0 & 63;

  auto load_tiles = [&](int buf, int k0) {
#if ASYNC_LDS
    __builtin_amdgcn_global_load_async_to_lds_b128(
        TO_GLB(A + (size_t)(m0 + ar0) * K + k0 + ac0), TO_LDS(&As[buf][ar0][ac0]), 0, 0);
    __builtin_amdgcn_global_load_async_to_lds_b128(
        TO_GLB(A + (size_t)(m0 + ar1) * K + k0 + ac1), TO_LDS(&As[buf][ar1][ac1]), 0, 0);
    __builtin_amdgcn_global_load_async_to_lds_b128(
        TO_GLB(W + (size_t)(k0 + br0) * N + n0 + bc0), TO_LDS(&Bs[buf][br0][bc0]), 0, 0);
#else
    *(uint4*)&As[buf][ar0][ac0] = *(const uint4*)(A + (size_t)(m0 + ar0) * K + k0 + ac0);
    *(uint4*)&As[buf][ar1][ac1] = *(const uint4*)(A + (size_t)(m0 + ar1) * K + k0 + ac1);
    *(uint4*)&Bs[buf][br0][bc0] = *(const uint4*)(W + (size_t)(k0 + br0) * N + n0 + bc0);
#endif
  };

  // prologue: fill buffer 0
  load_tiles(0, 0);
#if ASYNC_LDS
  __builtin_amdgcn_s_wait_asynccnt(0);
#endif
  __syncthreads();

  v8f acc[2][2] = {};
  int buf = 0;
  for (int k0 = 0; k0 < K; k0 += 32) {
    const int k1 = k0 + 32;
    if (k1 < K) load_tiles(buf ^ 1, k1);   // stream next tile while computing

    v16h af[2], bf[2];
#pragma unroll
    for (int v = 0; v < 8; ++v)
#pragma unroll
      for (int p = 0; p < 2; ++p) {
        int kk = frag_k(v, p, half);
        af[0][2 * v + p] = As[buf][wm * 32 + mr][kk];
        af[1][2 * v + p] = As[buf][wm * 32 + 16 + mr][kk];
        bf[0][2 * v + p] = Bs[buf][kk][wn * 32 + mr];
        bf[1][2 * v + p] = Bs[buf][kk][wn * 32 + 16 + mr];
      }
#pragma unroll
    for (int t = 0; t < 2; ++t)
#pragma unroll
      for (int u = 0; u < 2; ++u)
        acc[t][u] = wmma_f16(af[t], bf[u], acc[t][u]);

#if ASYNC_LDS
    if (k1 < K) __builtin_amdgcn_s_wait_asynccnt(0);
#endif
    __syncthreads();
    buf ^= 1;
  }

#pragma unroll
  for (int t = 0; t < 2; ++t)
#pragma unroll
    for (int u = 0; u < 2; ++u)
#pragma unroll
      for (int v = 0; v < 8; ++v) {
        int row = m0 + wm * 32 + t * 16 + v + 8 * half;
        int col = n0 + wn * 32 + u * 16 + mr;
        float val = acc[t][u][v];
        if (bias) val += bias[col];
        if (ACT == 1) val = gelu_f(val);
        if (OUT16) ((_Float16*)Cout)[(size_t)row * N + col] = (_Float16)val;
        else       ((float*)Cout)[(size_t)row * N + col] = val;
      }
}

// ---------------------------------------------------------------------------
// Fused attention for one (b, h, 16-query tile), f16 Q/K/V:
//   scores = Q*K^T * scale + bias, mask, softmax -> Pout (f32) ;  O = P*V (f32)
// attn_bias is read once and P written once (~1 GB/launch): use non-temporal
// hints so they do not evict resident weights/activations from the 192MB L2.
// ---------------------------------------------------------------------------
__global__ __launch_bounds__(256)
void attn_kernel(const _Float16* __restrict__ Q, const _Float16* __restrict__ Kmat,
                 const _Float16* __restrict__ V, const float* __restrict__ bias,
                 const int* __restrict__ mask, float* __restrict__ Pout,
                 float* __restrict__ O) {
  __shared__ float Ps[16][SEQ];   // 64 KB score strip for 16 queries

  const int qt   = blockIdx.x;           // 0..63
  const int h    = blockIdx.y;           // 0..15
  const int b    = blockIdx.z;           // 0..1
  const int tid  = threadIdx.x;
  const int lane = tid & 31;
  const int wv   = tid >> 5;
  const int half = lane >> 4;
  const int mr   = lane & 15;
  const int q0   = qt * 16;
  const float scale = 0.125f;            // 1/sqrt(64)

  const _Float16* Qb = Q    + (size_t)b * SEQ * DMODEL + (size_t)h * DK;
  const _Float16* Kb = Kmat + (size_t)b * SEQ * DMODEL + (size_t)h * DK;
  const _Float16* Vb = V    + (size_t)b * SEQ * DMODEL + (size_t)h * DK;

  // Q fragments are invariant across key tiles: load once
  v16h aq[2];
#pragma unroll
  for (int ks = 0; ks < 2; ++ks)
#pragma unroll
    for (int v = 0; v < 8; ++v)
#pragma unroll
      for (int p = 0; p < 2; ++p)
        aq[ks][2 * v + p] = Qb[(size_t)(q0 + mr) * DMODEL + ks * 32 + frag_k(v, p, half)];

  // ---- Phase 1: score strip via WMMA, 8 waves x 8 key-tiles each ----
  for (int kt = wv; kt < SEQ / 16; kt += 8) {
    const int kcol0 = kt * 16;
    v8f acc = {};
#pragma unroll
    for (int ks = 0; ks < 2; ++ks) {     // DK = 64 = 2 x 32
      v16h bf;
#pragma unroll
      for (int v = 0; v < 8; ++v)
#pragma unroll
        for (int p = 0; p < 2; ++p)
          bf[2 * v + p] = Kb[(size_t)(kcol0 + mr) * DMODEL + ks * 32 + frag_k(v, p, half)];
      acc = wmma_f16(aq[ks], bf, acc);
    }
#pragma unroll
    for (int v = 0; v < 8; ++v) {
      int qrow = v + 8 * half;
      int kcol = kcol0 + mr;
      float bv = __builtin_nontemporal_load(
          &bias[(((size_t)b * NHEAD + h) * SEQ + (q0 + qrow)) * SEQ + kcol]);
      float s = acc[v] * scale + bv;
      if (mask[b * SEQ + kcol] == 0) s = -9.0e15f;
      Ps[qrow][kcol] = s;
    }
  }
  __syncthreads();

  // ---- Phase 2: row softmax (each wave: 2 rows; 16 lanes per row) ----
  {
    const int row = 2 * wv + half;
    float mx = -3.0e38f;
    for (int j = mr; j < SEQ; j += 16) mx = fmaxf(mx, Ps[row][j]);
#pragma unroll
    for (int off = 8; off >= 1; off >>= 1) mx = fmaxf(mx, __shfl_xor(mx, off, 32));
    float sum = 0.0f;
    for (int j = mr; j < SEQ; j += 16) {
      float e = __expf(Ps[row][j] - mx);
      Ps[row][j] = e;
      sum += e;
    }
#pragma unroll
    for (int off = 8; off >= 1; off >>= 1) sum += __shfl_xor(sum, off, 32);
    const float inv = 1.0f / sum;
    float* prow = Pout + (((size_t)b * NHEAD + h) * SEQ + (q0 + row)) * (size_t)SEQ;
    for (int j = mr; j < SEQ; j += 16) {
      float pv = Ps[row][j] * inv;
      Ps[row][j] = pv;
      __builtin_nontemporal_store(pv, &prow[j]);   // streaming, never re-read
    }
  }
  __syncthreads();

  // ---- Phase 3: O(16x64) = P(16x1024) * V(1024x64), 4 waves ----
  if (wv < 4) {
    const int nt = wv;                   // 16-col slice of head dim
    v8f acc = {};
    for (int ks = 0; ks < SEQ / 32; ++ks) {
      v16h af, bf;
#pragma unroll
      for (int v = 0; v < 8; ++v)
#pragma unroll
        for (int p = 0; p < 2; ++p) {
          int kk = frag_k(v, p, half);
          af[2 * v + p] = (_Float16)Ps[mr][ks * 32 + kk];
          bf[2 * v + p] = Vb[(size_t)(ks * 32 + kk) * DMODEL + nt * 16 + mr];
        }
      acc = wmma_f16(af, bf, acc);
    }
#pragma unroll
    for (int v = 0; v < 8; ++v) {
      int qrow = v + 8 * half;
      O[((size_t)b * SEQ + q0 + qrow) * DMODEL + h * DK + nt * 16 + mr] = acc[v];
    }
  }
}

// ---------------------------------------------------------------------------
// out[row] = LayerNorm(a[row] + r[row]) * g + be   (one block per row)
// ---------------------------------------------------------------------------
__global__ __launch_bounds__(256)
void add_ln_kernel(const float* __restrict__ a, const float* __restrict__ r,
                   const float* __restrict__ g, const float* __restrict__ be,
                   float* __restrict__ out) {
  __shared__ float red[256];
  const int row = blockIdx.x;
  const int tid = threadIdx.x;
  const float* ap = a + (size_t)row * DMODEL;
  const float* rp = r + (size_t)row * DMODEL;

  float vals[4];
  float s = 0.0f;
#pragma unroll
  for (int i = 0; i < 4; ++i) {
    vals[i] = ap[tid + i * 256] + rp[tid + i * 256];
    s += vals[i];
  }
  red[tid] = s;
  __syncthreads();
  for (int st = 128; st > 0; st >>= 1) {
    if (tid < st) red[tid] += red[tid + st];
    __syncthreads();
  }
  const float mu = red[0] * (1.0f / DMODEL);
  __syncthreads();

  float vs = 0.0f;
#pragma unroll
  for (int i = 0; i < 4; ++i) {
    float d = vals[i] - mu;
    vs += d * d;
  }
  red[tid] = vs;
  __syncthreads();
  for (int st = 128; st > 0; st >>= 1) {
    if (tid < st) red[tid] += red[tid + st];
    __syncthreads();
  }
  const float inv = rsqrtf(red[0] * (1.0f / DMODEL) + 1e-6f);
#pragma unroll
  for (int i = 0; i < 4; ++i) {
    int d = tid + i * 256;
    out[(size_t)row * DMODEL + d] = (vals[i] - mu) * inv * g[d] + be[d];
  }
}

// ---------------------------------------------------------------------------
// host-side orchestration
// ---------------------------------------------------------------------------
extern "C" void kernel_launch(void* const* d_in, const int* in_sizes, int n_in,
                              void* d_out, int out_size, void* d_ws, size_t ws_size,
                              hipStream_t stream) {
  (void)in_sizes; (void)n_in; (void)out_size; (void)ws_size;

  const float* x_in   = (const float*)d_in[0];
  const int*   mask   = (const int*)  d_in[1];
  const float* abias  = (const float*)d_in[2];
  const float* Wq     = (const float*)d_in[3];
  const float* Wk     = (const float*)d_in[4];
  const float* Wv     = (const float*)d_in[5];
  const float* ln1_g  = (const float*)d_in[6];
  const float* ln1_b  = (const float*)d_in[7];
  const float* W1     = (const float*)d_in[8];
  const float* b1     = (const float*)d_in[9];
  const float* W2     = (const float*)d_in[10];
  const float* b2     = (const float*)d_in[11];
  const float* ln2_g  = (const float*)d_in[12];
  const float* ln2_b  = (const float*)d_in[13];

  float* out        = (float*)d_out;
  float* scores_out = out;                                            // [L,B,H,S,S]
  const size_t scores_layer = (size_t)BATCH * NHEAD * SEQ * SEQ;
  float* x_out      = out + (size_t)NLAYER * scores_layer;            // [B,S,D]

  // ---- workspace carve (f32 buffers first, then f16) ----
  char* wsb = (char*)d_ws;
  float* Ob = (float*)wsb; wsb += (size_t)MROWS * DMODEL * sizeof(float);
  float* Fb = (float*)wsb; wsb += (size_t)MROWS * DMODEL * sizeof(float);
  float* XA = (float*)wsb; wsb += (size_t)MROWS * DMODEL * sizeof(float);
  float* XB = (float*)wsb; wsb += (size_t)MROWS * DMODEL * sizeof(float);
  _Float16* Qh = (_Float16*)wsb; wsb += (size_t)MROWS * DMODEL * sizeof(_Float16);
  _Float16* Kh = (_Float16*)wsb; wsb += (size_t)MROWS * DMODEL * sizeof(_Float16);
  _Float16* Vh = (_Float16*)wsb; wsb += (size_t)MROWS * DMODEL * sizeof(_Float16);
  _Float16* Hh = (_Float16*)wsb; wsb += (size_t)MROWS * DFF    * sizeof(_Float16);
  _Float16* Xh = (_Float16*)wsb; wsb += (size_t)MROWS * DMODEL * sizeof(_Float16);
  _Float16* Wh = (_Float16*)wsb; wsb += (size_t)DMODEL * DFF   * sizeof(_Float16);

  const dim3 blk(256);
  const dim3 grid_dd(DMODEL / 64, MROWS / 128);   // N=1024
  const dim3 grid_df(DFF / 64,    MROWS / 128);   // N=2048
  const dim3 grid_at(SEQ / 16, NHEAD, BATCH);

  auto cvt = [&](const float* src, _Float16* dst, size_t n) {
    cvt_kernel<<<dim3((unsigned)(n / 2048)), blk, 0, stream>>>(src, dst);
  };

  const float* xcur = x_in;
  for (int n = 0; n < NLAYER; ++n) {
    const size_t wdd = (size_t)DMODEL * DMODEL;
    const size_t wdf = (size_t)DMODEL * DFF;

    // activations -> f16 once
    cvt(xcur, Xh, (size_t)MROWS * DMODEL);

    // QKV projections (f16 in, f16 out)
    cvt(Wq + (size_t)n * wdd, Wh, wdd);
    gemm16_kernel<0, 1><<<grid_dd, blk, 0, stream>>>(Xh, Wh, nullptr, Qh, MROWS, DMODEL, DMODEL);
    cvt(Wk + (size_t)n * wdd, Wh, wdd);
    gemm16_kernel<0, 1><<<grid_dd, blk, 0, stream>>>(Xh, Wh, nullptr, Kh, MROWS, DMODEL, DMODEL);
    cvt(Wv + (size_t)n * wdd, Wh, wdd);
    gemm16_kernel<0, 1><<<grid_dd, blk, 0, stream>>>(Xh, Wh, nullptr, Vh, MROWS, DMODEL, DMODEL);

    attn_kernel<<<grid_at, blk, 0, stream>>>(Qh, Kh, Vh, abias, mask,
                                             scores_out + (size_t)n * scores_layer, Ob);

    add_ln_kernel<<<dim3(MROWS), blk, 0, stream>>>(Ob, xcur,
        ln1_g + (size_t)n * DMODEL, ln1_b + (size_t)n * DMODEL, XA);

    // FFN: h = gelu(XA*W1 + b1) (f16 out), f = h*W2 + b2 (f32 out)
    cvt(XA, Xh, (size_t)MROWS * DMODEL);
    cvt(W1 + (size_t)n * wdf, Wh, wdf);
    gemm16_kernel<1, 1><<<grid_df, blk, 0, stream>>>(Xh, Wh, b1 + (size_t)n * DFF,
                                                     Hh, MROWS, DFF, DMODEL);
    cvt(W2 + (size_t)n * wdf, Wh, wdf);
    gemm16_kernel<0, 0><<<grid_dd, blk, 0, stream>>>(Hh, Wh, b2 + (size_t)n * DMODEL,
                                                     Fb, MROWS, DMODEL, DFF);

    float* xnext = (n == NLAYER - 1) ? x_out : XB;
    add_ln_kernel<<<dim3(MROWS), blk, 0, stream>>>(Fb, XA,
        ln2_g + (size_t)n * DMODEL, ln2_b + (size_t)n * DMODEL, xnext);
    xcur = xnext;
  }
}
